// MultiHeadSelfAttention_70884140253873
// MI455X (gfx1250) — compile-verified, matrix operand order checked by reading, not run on hardware
//
#include <hip/hip_runtime.h>

// MultiHeadSelfAttention forward, MI455X (gfx1250), bf16 WMMA path.
// B=2, S=2048, D_MODEL=1024, H=16, Dk=64.
// Compute-bound (~60 GFLOP, working set < L2): keep the WMMA pipe busy via
// register double-buffering so loads for step i+1 overlap WMMAs of step i.

#define D_MODEL 1024
#define N_HEADS 16
#define D_K     64
#define SEQ     2048
#define BATCH   2
#define BS      (BATCH * SEQ)   // 4096 rows

typedef __attribute__((ext_vector_type(16))) __bf16         v16bf;
typedef __attribute__((ext_vector_type(8)))  float          v8f;
typedef __attribute__((ext_vector_type(8)))  unsigned short v8u16;

union FragU { v16bf bf; v8u16 u[2]; };

__device__ __forceinline__ unsigned short f2bf(float f) {
  union { float f; unsigned int u; } a; a.f = f;
  unsigned int u = a.u;
  return (unsigned short)((u + 0x7FFFu + ((u >> 16) & 1u)) >> 16);  // RNE
}

// Load a 16x32 bf16 tile (row-major, stride `ld` elements) into the WMMA
// A-frag layout (B-operands are loaded via their transpose => same layout):
//   lane L (L<16):  row M=L,  dwords = K{0..7} then K{16..23}
//   lane L+16:      row M=L,  dwords = K{8..15} then K{24..31}
__device__ __forceinline__ v16bf load_frag(const unsigned short* p, int ld) {
  int lane = threadIdx.x & 31;
  int row  = lane & 15;
  int kg   = lane >> 4;
  const unsigned short* q = p + row * ld + kg * 8;
  FragU f;
  f.u[0] = *(const v8u16*)(q);
  f.u[1] = *(const v8u16*)(q + 16);
  return f.bf;
}

__device__ __forceinline__ v8f wmma_bf16(v16bf a, v16bf b, v8f c) {
  // (neg_a, A, neg_b, B, c_mod, C, reuse_a, reuse_b)
  return __builtin_amdgcn_wmma_f32_16x16x32_bf16(false, a, false, b, (short)0, c,
                                                 false, false);
}

// ---------------- prep: f32 -> bf16 conversions ----------------

__global__ __launch_bounds__(256) void cvt_x_kernel(const float* __restrict__ x,
                                                    unsigned short* __restrict__ xb) {
  int i = blockIdx.x * 256 + threadIdx.x;
  xb[i] = f2bf(x[i]);
}

// Wt[z][n][k] = W[z][k][n]  (bf16) so GEMM B-operands are contiguous rows.
__global__ __launch_bounds__(256) void cvt_w_kernel(const float* __restrict__ Wq,
                                                    const float* __restrict__ Wk,
                                                    const float* __restrict__ Wv,
                                                    unsigned short* __restrict__ wt) {
  int z = blockIdx.z;
  const float* W = (z == 0) ? Wq : (z == 1) ? Wk : Wv;
  int i = blockIdx.x * 256 + threadIdx.x;     // 0 .. 1M-1, coalesced read
  int k = i >> 10, n = i & 1023;
  wt[(size_t)z * D_MODEL * D_MODEL + (size_t)n * D_MODEL + k] =
      f2bf(W[(size_t)k * D_MODEL + n]);
}

// ---------------- QKV projection GEMM ----------------
// One wave computes a 16x64 tile of (x @ W + b), software-pipelined:
// fragments for K-step i+1 are in flight while step i's 4 WMMAs execute.
// Q,K stored [B,H,S,Dk]; V stored transposed [B,H,Dk,S].

__global__ __launch_bounds__(256)
void qkv_gemm_kernel(const unsigned short* __restrict__ xb,
                     const unsigned short* __restrict__ wt,
                     const float* __restrict__ bq, const float* __restrict__ bk,
                     const float* __restrict__ bv,
                     unsigned short* __restrict__ qb,
                     unsigned short* __restrict__ kb,
                     unsigned short* __restrict__ vb) {
  int z = blockIdx.z;  // 0=Q 1=K 2=V (uniform per block)
  const unsigned short* W = wt + (size_t)z * D_MODEL * D_MODEL;
  const float* bias = (z == 0) ? bq : (z == 1) ? bk : bv;

  int wave = threadIdx.x >> 5;
  int task = blockIdx.x * 8 + wave;   // 4096 tasks = 256 m-tiles * 16 n-groups
  int mt = task >> 4;                 // 16-row tile of x
  int ng = task & 15;                 // 64-wide n group == head index
  int lane  = threadIdx.x & 31;
  int nlane = lane & 15;
  int g     = lane >> 4;

  const unsigned short* A  = xb + (size_t)(mt * 16) * D_MODEL;
  const unsigned short* B0 = W + (size_t)(ng * 64 +  0) * D_MODEL;
  const unsigned short* B1 = W + (size_t)(ng * 64 + 16) * D_MODEL;
  const unsigned short* B2 = W + (size_t)(ng * 64 + 32) * D_MODEL;
  const unsigned short* B3 = W + (size_t)(ng * 64 + 48) * D_MODEL;

  v8f acc[4] = {};

  // prologue: fragments for k = 0
  v16bf af  = load_frag(A,  D_MODEL);
  v16bf bf0 = load_frag(B0, D_MODEL);
  v16bf bf1 = load_frag(B1, D_MODEL);
  v16bf bf2 = load_frag(B2, D_MODEL);
  v16bf bf3 = load_frag(B3, D_MODEL);

  for (int k = 0; k < D_MODEL - 32; k += 32) {
    // issue next K-step's loads first; they retire while WMMAs run
    int kn = k + 32;
    v16bf afn  = load_frag(A  + kn, D_MODEL);
    v16bf bn0  = load_frag(B0 + kn, D_MODEL);
    v16bf bn1  = load_frag(B1 + kn, D_MODEL);
    v16bf bn2  = load_frag(B2 + kn, D_MODEL);
    v16bf bn3  = load_frag(B3 + kn, D_MODEL);

    acc[0] = wmma_bf16(af, bf0, acc[0]);
    acc[1] = wmma_bf16(af, bf1, acc[1]);
    acc[2] = wmma_bf16(af, bf2, acc[2]);
    acc[3] = wmma_bf16(af, bf3, acc[3]);

    af = afn; bf0 = bn0; bf1 = bn1; bf2 = bn2; bf3 = bn3;
  }
  // epilogue K-step (k = D_MODEL-32)
  acc[0] = wmma_bf16(af, bf0, acc[0]);
  acc[1] = wmma_bf16(af, bf1, acc[1]);
  acc[2] = wmma_bf16(af, bf2, acc[2]);
  acc[3] = wmma_bf16(af, bf3, acc[3]);

#pragma unroll
  for (int j = 0; j < 4; ++j) {
    int n   = ng * 64 + j * 16 + nlane;
    int dk  = j * 16 + nlane;
    float bias_n = bias[n];
#pragma unroll
    for (int r = 0; r < 8; ++r) {
      int m = mt * 16 + r + 8 * g;        // C-frag: M = vgpr + 8*(lane/16)
      int b = m >> 11, s = m & 2047;
      unsigned short val = f2bf(acc[j][r] + bias_n);
      if (z == 2) {
        vb[(((size_t)(b * N_HEADS + ng)) * D_K + dk) * SEQ + s] = val;   // V^T
      } else {
        unsigned short* dst = (z == 0) ? qb : kb;
        dst[(((size_t)(b * N_HEADS + ng)) * SEQ + s) * D_K + dk] = val;
      }
    }
  }
}

// ---------------- fused attention ----------------
// One wave per (batch*head, 16-query tile). Stream 32-key blocks:
//   S = Q K^T / 8 ; E = exp(S) ; ctx += E V ; rowsum += E @ ones (via WMMA)
// Final: out = ctx / (rowsum + 1e-8). K-frags double-buffered across blocks;
// V-frags loaded before exp/LDS staging so their latency hides behind VALU.

__global__ __launch_bounds__(256)
void attn_kernel(const unsigned short* __restrict__ qb,
                 const unsigned short* __restrict__ kb,
                 const unsigned short* __restrict__ vb,
                 float* __restrict__ out) {
  __shared__ __align__(16) unsigned short eTile[8][16 * 32];  // per-wave E stage

  int wave = threadIdx.x >> 5;
  int lane = threadIdx.x & 31;
  int task = blockIdx.x * 8 + wave;   // 4096 tasks = 32 bh * 128 q-tiles
  int bh = task >> 7;
  int qt = task & 127;
  int b = bh >> 4, h = bh & 15;

  const unsigned short* Q  = qb + ((size_t)bh * SEQ + qt * 16) * D_K;
  const unsigned short* K  = kb + (size_t)bh * SEQ * D_K;
  const unsigned short* Vt = vb + (size_t)bh * D_K * SEQ;

  v16bf qf0 = load_frag(Q, D_K);        // queries, d = 0..31
  v16bf qf1 = load_frag(Q + 32, D_K);   // queries, d = 32..63

  FragU onesU;
#pragma unroll
  for (int i = 0; i < 8; ++i) { onesU.u[0][i] = 0x3F80; onesU.u[1][i] = 0x3F80; }
  v16bf ones = onesU.bf;

  v8f ctx[4] = {};
  v8f rs = {};

  int nlane = lane & 15;
  int g     = lane >> 4;
  unsigned short* E = &eTile[wave][0];

  // prologue: K-frags for block 0
  v16bf kf0 = load_frag(K, D_K);
  v16bf kf1 = load_frag(K + 32, D_K);
  v16bf kf2 = load_frag(K + 16 * D_K, D_K);
  v16bf kf3 = load_frag(K + 16 * D_K + 32, D_K);

  for (int kblk = 0; kblk < SEQ / 32; ++kblk) {
    int kbase = kblk * 32;

    // V-frags for this block: independent of the score WMMAs, issue early
    v16bf vf0 = load_frag(Vt + kbase, SEQ);
    v16bf vf1 = load_frag(Vt + (size_t)16 * SEQ + kbase, SEQ);
    v16bf vf2 = load_frag(Vt + (size_t)32 * SEQ + kbase, SEQ);
    v16bf vf3 = load_frag(Vt + (size_t)48 * SEQ + kbase, SEQ);
    __builtin_prefetch(Vt + kbase + 32, 0, 0);  // next V block -> cache

    v8f s0 = {}, s1 = {};
    s0 = wmma_bf16(qf0, kf0, s0);   // keys 0..15, d lo
    s0 = wmma_bf16(qf1, kf1, s0);   // keys 0..15, d hi
    s1 = wmma_bf16(qf0, kf2, s1);   // keys 16..31, d lo
    s1 = wmma_bf16(qf1, kf3, s1);   // keys 16..31, d hi

    // preload next block's K-frags (clamped on the last block)
    int nb = (kbase + 32 < SEQ) ? kbase + 32 : kbase;
    const unsigned short* Kn = K + (size_t)nb * D_K;
    kf0 = load_frag(Kn, D_K);
    kf1 = load_frag(Kn + 32, D_K);
    kf2 = load_frag(Kn + 16 * D_K, D_K);
    kf3 = load_frag(Kn + 16 * D_K + 32, D_K);

    // E = exp(S / sqrt(64)); restage C-frag -> A-frag via wave-private LDS
#pragma unroll
    for (int r = 0; r < 8; ++r) {
      int m = r + 8 * g;
      E[m * 32 + nlane]      = f2bf(__expf(s0[r] * 0.125f));
      E[m * 32 + 16 + nlane] = f2bf(__expf(s1[r] * 0.125f));
    }
    v16bf ef = load_frag(E, 32);

    ctx[0] = wmma_bf16(ef, vf0, ctx[0]);
    ctx[1] = wmma_bf16(ef, vf1, ctx[1]);
    ctx[2] = wmma_bf16(ef, vf2, ctx[2]);
    ctx[3] = wmma_bf16(ef, vf3, ctx[3]);
    rs     = wmma_bf16(ef, ones, rs);   // every column = row-sum of E
  }

#pragma unroll
  for (int j = 0; j < 4; ++j) {
    int d = j * 16 + nlane;
#pragma unroll
    for (int r = 0; r < 8; ++r) {
      int m = r + 8 * g;
      int s = qt * 16 + m;
      out[((size_t)b * SEQ + s) * D_MODEL + h * D_K + d] =
          ctx[j][r] / (rs[r] + 1e-8f);
    }
  }
}

// ---------------- launch ----------------

extern "C" void kernel_launch(void* const* d_in, const int* in_sizes, int n_in,
                              void* d_out, int out_size, void* d_ws, size_t ws_size,
                              hipStream_t stream) {
  const float* x  = (const float*)d_in[0];
  const float* Wq = (const float*)d_in[1];
  const float* bq = (const float*)d_in[2];
  const float* Wk = (const float*)d_in[3];
  const float* bk = (const float*)d_in[4];
  const float* Wv = (const float*)d_in[5];
  const float* bv = (const float*)d_in[6];
  float* out = (float*)d_out;

  // workspace layout (bf16 elements): xb | Wt(x3) | Q | K | V^T  (~40 MB)
  unsigned short* xb   = (unsigned short*)d_ws;
  unsigned short* wt   = xb + (size_t)BS * D_MODEL;
  unsigned short* qbuf = wt + (size_t)3 * D_MODEL * D_MODEL;
  unsigned short* kbuf = qbuf + (size_t)BS * D_MODEL;
  unsigned short* vbuf = kbuf + (size_t)BS * D_MODEL;

  cvt_x_kernel<<<(BS * D_MODEL) / 256, 256, 0, stream>>>(x, xb);
  cvt_w_kernel<<<dim3(D_MODEL * D_MODEL / 256, 1, 3), 256, 0, stream>>>(Wq, Wk, Wv, wt);
  qkv_gemm_kernel<<<dim3(512, 1, 3), 256, 0, stream>>>(xb, wt, bq, bk, bv,
                                                       qbuf, kbuf, vbuf);
  attn_kernel<<<dim3(512), 256, 0, stream>>>(qbuf, kbuf, vbuf, out);
}
